// MHA_NAIVE_15015205667367
// MI455X (gfx1250) — matrix operand order repaired and checked by reading, not compiled, not run on hardware
//
#include <hip/hip_runtime.h>

// Problem constants (from reference)
#define DIM_  1024
#define NH_   16
#define HD_   64
#define B_    2
#define S_    2048
#define MTOT_ (B_ * S_)    // 4096 rows total (batch folded into M)
#define E3_   (3 * HD_)    // 192 qkv columns per head

typedef __attribute__((ext_vector_type(16))) _Float16 v16h;
typedef __attribute__((ext_vector_type(8)))  float    v8f;

union V16H { v16h v; unsigned u[8]; };

__device__ __forceinline__ v8f wmma16(v16h a, v16h b, v8f c) {
  // D = A(16x32 f16) * B(32x16 f16) + C(16x16 f32)
  return __builtin_amdgcn_wmma_f32_16x16x32_f16(false, a, false, b, (short)0, c,
                                                false, false);
}

// --- A operand: 16x32 f16 tile from row-major [M, K] source (stride ld elems).
// A layout (matching B): lanes 0-15 hold M=lane, K in {0..15};
// lanes 16-31 hold M=lane-16, K in {16..31}; 2 f16 per VGPR.
__device__ __forceinline__ v16h load_tileA(const _Float16* __restrict__ src, int ld) {
  const int lane = threadIdx.x & 31;
  const int kh = (lane >> 4) << 4;  // 0 or 16
  const _Float16* p = src + (size_t)(lane & 15) * ld + kh;
  V16H t;
#pragma unroll
  for (int i = 0; i < 8; ++i) {
    t.u[i]     = *(const unsigned*)(p + 2 * i);        // K = kh+2i, kh+2i+1
  }
  return t.v;
}

// --- B operand: 32x16 f16 tile (KxN); source is row-major [N, K] (i.e. B^T).
// B layout used here (contiguous k, same as A): lanes 0-15 hold N=lane, K=0..15;
// lanes 16-31 hold N=lane-16, K=16..31; VGPR v holds K=2v,2v+1.
// => per lane: 32 contiguous bytes starting at (N*ld + 16*(lane>>4)).
__device__ __forceinline__ v16h load_tileB(const _Float16* __restrict__ src, int ld) {
  const int lane = threadIdx.x & 31;
  const _Float16* p = src + (size_t)(lane & 15) * ld + ((lane >> 4) << 4);
  V16H t;
#pragma unroll
  for (int i = 0; i < 8; ++i) t.u[i] = *(const unsigned*)(p + 2 * i);
  return t.v;
}

// ---------------- conversion / transpose staging kernels ----------------

__global__ void cvt_f16_kernel(const float* __restrict__ src,
                               _Float16* __restrict__ dst, int n) {
  int i = blockIdx.x * blockDim.x + threadIdx.x;
  if (i < n) dst[i] = (_Float16)src[i];
}

// dst[z][c][r] = src[z][r][c], converted to f16. grid.z batches slices.
__global__ void transpose_cvt_kernel(const float* __restrict__ src,
                                     _Float16* __restrict__ dst,
                                     int rows, int cols) {
  const size_t total = (size_t)rows * cols;
  size_t i = (size_t)blockIdx.x * blockDim.x + threadIdx.x;
  src += (size_t)blockIdx.z * total;
  dst += (size_t)blockIdx.z * total;
  if (i < total) {
    int r = (int)(i / cols);
    int c = (int)(i % cols);
    dst[(size_t)c * rows + r] = (_Float16)src[i];
  }
}

// ---------------- QKV projection GEMM (register-blocked) ----------------
// Each wave computes a 64x32 C region (4 M-subtiles x 2 N-subtiles): 8 WMMAs
// per k-step fed by 4 A-tile + 2 B-tile loads (~21 FLOP/byte of L2 traffic).
// grid: (MTOT/64, 1, NH), block 192 (6 waves cover all 192 N columns).
// xb:  [MTOT, DIM] f16      wt: [NH, 192, DIM] f16 (pre-transposed weights)
// Outputs: Qh,Kh [B,H,S,HD] f16;  Vth [B,H,HD,S] f16 (V stored transposed).
__global__ void qkv_gemm_kernel(const _Float16* __restrict__ xb,
                                const _Float16* __restrict__ wt,
                                const float* __restrict__ bq,
                                _Float16* __restrict__ Qh,
                                _Float16* __restrict__ Kh,
                                _Float16* __restrict__ Vth) {
  const int wave = threadIdx.x >> 5;
  const int lane = threadIdx.x & 31;
  const int m0 = blockIdx.x * 64;
  const int n0 = wave * 32;                  // 0..160
  const int h  = blockIdx.z;
  const _Float16* A  = xb + (size_t)m0 * DIM_;
  const _Float16* Bt = wt + ((size_t)h * E3_ + n0) * DIM_;

  v8f zero = {};
  v8f acc[4][2];
#pragma unroll
  for (int mi = 0; mi < 4; ++mi)
#pragma unroll
    for (int nj = 0; nj < 2; ++nj) acc[mi][nj] = zero;

  for (int k0 = 0; k0 < DIM_; k0 += 32) {
    v16h a[4], b[2];
#pragma unroll
    for (int mi = 0; mi < 4; ++mi)
      a[mi] = load_tileA(A + (size_t)(mi * 16) * DIM_ + k0, DIM_);
#pragma unroll
    for (int nj = 0; nj < 2; ++nj)
      b[nj] = load_tileB(Bt + (size_t)(nj * 16) * DIM_ + k0, DIM_);
#pragma unroll
    for (int mi = 0; mi < 4; ++mi)
#pragma unroll
      for (int nj = 0; nj < 2; ++nj)
        acc[mi][nj] = wmma16(a[mi], b[nj], acc[mi][nj]);
  }

  const int rowhi = (lane >> 4) << 3;
#pragma unroll
  for (int nj = 0; nj < 2; ++nj) {
    const int col = n0 + nj * 16 + (lane & 15);
    const float bias = bq[h * E3_ + col];
#pragma unroll
    for (int mi = 0; mi < 4; ++mi) {
#pragma unroll
      for (int vg = 0; vg < 8; ++vg) {
        const int row = m0 + mi * 16 + vg + rowhi;  // [0, MTOT)
        const int bb = row >> 11;                   // / S_
        const int s  = row & (S_ - 1);
        const float val = acc[mi][nj][vg] + bias;
        const size_t head = (size_t)bb * NH_ + h;
        if (col < HD_)
          Qh[(head * S_ + s) * HD_ + col] = (_Float16)val;
        else if (col < 2 * HD_)
          Kh[(head * S_ + s) * HD_ + (col - HD_)] = (_Float16)val;
        else
          Vth[(head * HD_ + (col - 2 * HD_)) * S_ + s] = (_Float16)val;
      }
    }
  }
}

// ---------------- flash attention (causal) ----------------
// grid: (S/16, NH, B), block 32 (one wave per 16-query tile).
__global__ void attn_kernel(const _Float16* __restrict__ Qh,
                            const _Float16* __restrict__ Kh,
                            const _Float16* __restrict__ Vth,
                            _Float16* __restrict__ Yh) {
  __shared__ _Float16 lds[16 * 32];  // P tile, rows x 32 key cols
  const int lane = threadIdx.x & 31;
  const int q0 = blockIdx.x * 16;
  const int h  = blockIdx.y;
  const int b  = blockIdx.z;
  const size_t head = (size_t)b * NH_ + h;
  const _Float16* Qp = Qh + (head * S_ + q0) * HD_;
  const _Float16* Kp = Kh + head * S_ * HD_;
  const _Float16* Vp = Vth + head * HD_ * S_;

  const v16h aq0 = load_tileA(Qp, HD_);        // K(=d) 0..31
  const v16h aq1 = load_tileA(Qp + 32, HD_);   // K(=d) 32..63

  const int col = lane & 15;
  const int rowhi = (lane >> 4) << 3;          // this lane covers rows rowhi..rowhi+7

  float mrow[8], lrow[8];
  v8f zero = {};
  v8f o[4] = {zero, zero, zero, zero};         // 16 x 64 f32 accumulator
#pragma unroll
  for (int vg = 0; vg < 8; ++vg) { mrow[vg] = -3.0e38f; lrow[vg] = 0.f; }

  const float scale = 0.125f;                  // 1/sqrt(HD)

  for (int t0 = 0; t0 <= q0 + 15; t0 += 32) {
    // ---- score tiles S[16 x 32] = Q(16x64) * K^T ----
    v8f sc0 = zero, sc1 = zero;
    {
      v16h bk = load_tileB(Kp + (size_t)t0 * HD_, HD_);
      sc0 = wmma16(aq0, bk, sc0);
      bk = load_tileB(Kp + (size_t)t0 * HD_ + 32, HD_);
      sc0 = wmma16(aq1, bk, sc0);
      bk = load_tileB(Kp + (size_t)(t0 + 16) * HD_, HD_);
      sc1 = wmma16(aq0, bk, sc1);
      bk = load_tileB(Kp + (size_t)(t0 + 16) * HD_ + 32, HD_);
      sc1 = wmma16(aq1, bk, sc1);
    }
    // ---- scale + causal mask + per-row running max ----
    float mnew[8];
#pragma unroll
    for (int vg = 0; vg < 8; ++vg) {
      const int row = q0 + vg + rowhi;
      float v0 = sc0[vg] * scale;
      float v1 = sc1[vg] * scale;
      v0 = (t0 + col      <= row) ? v0 : -3.0e38f;
      v1 = (t0 + 16 + col <= row) ? v1 : -3.0e38f;
      sc0[vg] = v0; sc1[vg] = v1;
      mnew[vg] = fmaxf(mrow[vg], fmaxf(v0, v1));
    }
    // row max: reduce across the 16 lanes of each half-wave (xor < 16)
#pragma unroll
    for (int sft = 1; sft < 16; sft <<= 1) {
#pragma unroll
      for (int vg = 0; vg < 8; ++vg)
        mnew[vg] = fmaxf(mnew[vg], __shfl_xor(mnew[vg], sft, 32));
    }
    // ---- online softmax: rescale, exp, accumulate l, stage P in LDS ----
#pragma unroll
    for (int vg = 0; vg < 8; ++vg) {
      const float corr = __expf(mrow[vg] - mnew[vg]);
      mrow[vg] = mnew[vg];
      lrow[vg] *= corr;
      o[0][vg] *= corr; o[1][vg] *= corr; o[2][vg] *= corr; o[3][vg] *= corr;
      const float p0 = __expf(sc0[vg] - mnew[vg]);
      const float p1 = __expf(sc1[vg] - mnew[vg]);
      const int lr = vg + rowhi;
      lds[lr * 32 + col]      = (_Float16)p0;
      lds[lr * 32 + 16 + col] = (_Float16)p1;
      float ps = p0 + p1;
#pragma unroll
      for (int sft = 1; sft < 16; sft <<= 1) ps += __shfl_xor(ps, sft, 32);
      lrow[vg] += ps;
    }
    __syncthreads();  // single-wave block: orders LDS writes vs. reads
    // ---- O += P(16x32) * V(32x64): A from LDS, B from transposed V ----
    const v16h pa = load_tileA(lds, 32);
#pragma unroll
    for (int n = 0; n < 4; ++n) {
      const v16h bv = load_tileB(Vp + (size_t)(n * 16) * S_ + t0, S_);
      o[n] = wmma16(pa, bv, o[n]);
    }
    __syncthreads();
  }
  // ---- finalize: O / l, store heads-concatenated [B, S, H*HD] f16 ----
#pragma unroll
  for (int vg = 0; vg < 8; ++vg) {
    const float inv = 1.0f / lrow[vg];
    const int row = q0 + vg + rowhi;
    _Float16* yp = Yh + ((size_t)b * S_ + row) * DIM_ + h * HD_;
#pragma unroll
    for (int n = 0; n < 4; ++n)
      yp[n * 16 + col] = (_Float16)(o[n][vg] * inv);
  }
}

// ---------------- output projection GEMM (register-blocked) ----------------
// Each wave computes 64x32 (4 M x 2 N subtiles). Block = 8 waves -> 256 N cols.
// grid: (MTOT/64, DIM/256). out is fp32.
__global__ void out_gemm_kernel(const _Float16* __restrict__ Yh,
                                const _Float16* __restrict__ wt,  // [N, K] f16
                                const float* __restrict__ bo,
                                float* __restrict__ out) {
  const int wave = threadIdx.x >> 5;
  const int lane = threadIdx.x & 31;
  const int m0 = blockIdx.x * 64;
  const int n0 = blockIdx.y * 256 + wave * 32;
  const _Float16* A  = Yh + (size_t)m0 * DIM_;
  const _Float16* Bt = wt + (size_t)n0 * DIM_;

  v8f zero = {};
  v8f acc[4][2];
#pragma unroll
  for (int mi = 0; mi < 4; ++mi)
#pragma unroll
    for (int nj = 0; nj < 2; ++nj) acc[mi][nj] = zero;

  for (int k0 = 0; k0 < DIM_; k0 += 32) {
    v16h a[4], b[2];
#pragma unroll
    for (int mi = 0; mi < 4; ++mi)
      a[mi] = load_tileA(A + (size_t)(mi * 16) * DIM_ + k0, DIM_);
#pragma unroll
    for (int nj = 0; nj < 2; ++nj)
      b[nj] = load_tileB(Bt + (size_t)(nj * 16) * DIM_ + k0, DIM_);
#pragma unroll
    for (int mi = 0; mi < 4; ++mi)
#pragma unroll
      for (int nj = 0; nj < 2; ++nj)
        acc[mi][nj] = wmma16(a[mi], b[nj], acc[mi][nj]);
  }

  const int rowhi = (lane >> 4) << 3;
#pragma unroll
  for (int nj = 0; nj < 2; ++nj) {
    const int col = n0 + nj * 16 + (lane & 15);
    const float bias = bo[col];
#pragma unroll
    for (int mi = 0; mi < 4; ++mi)
#pragma unroll
      for (int vg = 0; vg < 8; ++vg) {
        const int row = m0 + mi * 16 + vg + rowhi;
        out[(size_t)row * DIM_ + col] = acc[mi][nj][vg] + bias;
      }
  }
}

// ---------------- launcher ----------------
extern "C" void kernel_launch(void* const* d_in, const int* in_sizes, int n_in,
                              void* d_out, int out_size, void* d_ws, size_t ws_size,
                              hipStream_t stream) {
  const float* x    = (const float*)d_in[0];  // [B, S, DIM]
  const float* Wqkv = (const float*)d_in[1];  // [H, DIM, 192]
  const float* bqkv = (const float*)d_in[2];  // [H, 192]
  const float* Wout = (const float*)d_in[3];  // [DIM, DIM]
  const float* bout = (const float*)d_in[4];  // [DIM]
  float* out = (float*)d_out;                 // [B, S, DIM] fp32

  char* ws = (char*)d_ws;
  size_t off = 0;
  auto alloc16 = [&](size_t elems) {
    _Float16* p = (_Float16*)(ws + off);
    off += ((elems * sizeof(_Float16)) + 255) & ~(size_t)255;
    return p;
  };
  _Float16* xb    = alloc16((size_t)MTOT_ * DIM_);        // x in f16
  _Float16* wqkvt = alloc16((size_t)NH_ * E3_ * DIM_);    // Wqkv^T f16 [H,192,DIM]
  _Float16* woutt = alloc16((size_t)DIM_ * DIM_);         // Wout^T f16 [N,K]
  _Float16* Qh    = alloc16((size_t)B_ * NH_ * S_ * HD_); // [B,H,S,HD]
  _Float16* Kh    = alloc16((size_t)B_ * NH_ * S_ * HD_); // [B,H,S,HD]
  _Float16* Vth   = alloc16((size_t)B_ * NH_ * HD_ * S_); // [B,H,HD,S]
  _Float16* Yh    = alloc16((size_t)MTOT_ * DIM_);        // attn out f16
  (void)ws_size; (void)in_sizes; (void)n_in; (void)out_size;

  cvt_f16_kernel<<<(MTOT_ * DIM_) / 256, 256, 0, stream>>>(x, xb, MTOT_ * DIM_);
  transpose_cvt_kernel<<<dim3((DIM_ * E3_) / 256, 1, NH_), 256, 0, stream>>>(
      Wqkv, wqkvt, DIM_, E3_);
  transpose_cvt_kernel<<<dim3((DIM_ * DIM_) / 256, 1, 1), 256, 0, stream>>>(
      Wout, woutt, DIM_, DIM_);

  qkv_gemm_kernel<<<dim3(MTOT_ / 64, 1, NH_), 192, 0, stream>>>(
      xb, wqkvt, bqkv, Qh, Kh, Vth);

  attn_kernel<<<dim3(S_ / 16, NH_, B_), 32, 0, stream>>>(Qh, Kh, Vth, Yh);

  out_gemm_kernel<<<dim3(MTOT_ / 64, DIM_ / 256, 1), 256, 0, stream>>>(
      Yh, woutt, bout, out);
}
